// LocalDownsample_12446815224156
// MI455X (gfx1250) — matrix operand order repaired and checked by reading, not compiled, not run on hardware
//
#include <hip/hip_runtime.h>

// ---------------------------------------------------------------------------
// LocalDownsample (segment mean) as an on-the-fly one-hot f16 WMMA GEMM.
//   B=8, T=4096, C=512, N=512 (fixed by the reference).
// Memory-optimal tiling: grid (C/64, B); each block reads its X slice once.
// X is streamed global->LDS with the Tensor Data Mover (double-buffered),
// converted to f16, and consumed by v_wmma_f32_16x16x32_f16.
// ---------------------------------------------------------------------------

typedef __attribute__((ext_vector_type(16))) _Float16     v16h;
typedef __attribute__((ext_vector_type(8)))  _Float16     v8h;
typedef __attribute__((ext_vector_type(8)))  float        v8f;
typedef __attribute__((ext_vector_type(4)))  unsigned int u32x4;
typedef __attribute__((ext_vector_type(8)))  int          i32x8;
typedef __attribute__((ext_vector_type(4)))  int          i32x4;

#define B_DIM 8
#define T_DIM 4096
#define C_DIM 512
#define N_DIM 512
#define KST   128   // K rows (t values) per LDS stage
#define CT    64    // C columns per block
#define PITCH 136   // halves per c-row in the f16 LDS tile (pad vs 128)

// LDS layout (dynamic shared only, so the TDM lds_addr offsets are known):
//  [0, 65536)              : float stage[2][KST][CT]   (TDM destination)
//  [65536, 65536+17408)    : _Float16 Xh[CT][PITCH]    ([c][k], 16B-aligned rows)
//  [82944, 82944+512)      : uint wcol[KST]            (scale_h<<16 | (region-1))
#define SM_STAGE   0
#define SM_XH      (2 * KST * CT * 4)
#define SM_WCOL    (SM_XH + CT * PITCH * 2)
#define SM_TOTAL   (SM_WCOL + KST * 4)

union Frag16 { v16h v; v8h h[2]; unsigned short u[16]; };

// ---------------------------------------------------------------------------
// Kernel 1: per-(b,n) counts -> f16 inverse counts in workspace.
// ---------------------------------------------------------------------------
__global__ void __launch_bounds__(256, 1)
seg_count_kernel(const long long* __restrict__ regions,
                 unsigned short* __restrict__ scale_h)
{
    __shared__ int cnt[N_DIM];
    const int b = blockIdx.x, tid = threadIdx.x;
    for (int n = tid; n < N_DIM; n += 256) cnt[n] = 0;
    __syncthreads();
    for (int t = tid; t < T_DIM; t += 256) {
        int r = (int)regions[(size_t)b * T_DIM + t] - 1;   // in [0, 511]
        atomicAdd(&cnt[r], 1);
    }
    __syncthreads();
    for (int n = tid; n < N_DIM; n += 256) {
        int c = cnt[n];
        _Float16 s = (c > 0) ? (_Float16)(1.0f / (float)c) : (_Float16)0.0f;
        scale_h[b * N_DIM + n] = __builtin_bit_cast(unsigned short, s);
    }
}

// ---------------------------------------------------------------------------
// TDM issue: 2D tile (tile_dim0=64 elems along C, tile_dim1=128 rows along T),
// 4-byte elements, row stride = C_DIM. Descriptor per CDNA5 ISA D# groups.
// This toolchain exposes the 6-arg builtin: groups 0..3 + extra payload + cpol.
// ---------------------------------------------------------------------------
__device__ __forceinline__ void tdm_load_stage(unsigned lds_off, const float* gptr)
{
    unsigned long long ga = (unsigned long long)(uintptr_t)gptr;
    u32x4 g0;
    g0[0] = 1u;                                              // count=1, user mode
    g0[1] = lds_off;                                         // lds_addr (bytes)
    g0[2] = (unsigned)(ga & 0xFFFFFFFFu);                    // global_addr lo
    g0[3] = (unsigned)((ga >> 32) & 0x01FFFFFFu) | (2u << 30); // addr hi | type=2
    i32x8 g1;
    g1[0] = (int)(2u << 16);          // workgroup_mask=0, data_size=2 (4 bytes)
    g1[1] = (int)(64u << 16);         // tensor_dim0 lo16 = 64  (== tile: no OOB)
    g1[2] = (int)(128u << 16);        // tensor_dim0 hi=0, tensor_dim1 lo16 = 128
    g1[3] = (int)(64u << 16);         // tensor_dim1 hi=0, tile_dim0 = 64
    g1[4] = (int)128;                 // tile_dim1 = 128, tile_dim2 = 0
    g1[5] = (int)C_DIM;               // tensor_dim0_stride lo32 = 512 elements
    g1[6] = 0;                        // stride hi, dim1_stride lo
    g1[7] = 0;
    i32x4 z4 = (i32x4)0;              // groups 2/3: tile_dim3/4 = 0 (unused)
    i32x8 z8 = (i32x8)0;              // extra descriptor payload: unused
    __builtin_amdgcn_tensor_load_to_lds(g0, g1, z4, z4, z8, 0);
}

// ---------------------------------------------------------------------------
// Kernel 2: out[b, 0:512, c0:c0+64] = W(b) @ X(b)[:, c0:c0+64] via WMMA.
// Block: 256 threads (8 waves). Wave w owns regions [64w, 64w+64).
// ---------------------------------------------------------------------------
__global__ void __launch_bounds__(256, 1)
seg_gemm_kernel(const float* __restrict__ x,
                const long long* __restrict__ regions,
                const unsigned short* __restrict__ scale_h,
                float* __restrict__ out)
{
    extern __shared__ char smem[];
    float*        stage = (float*)(smem + SM_STAGE);   // [2][KST*CT]
    _Float16*     Xh    = (_Float16*)(smem + SM_XH);   // [CT][PITCH]
    unsigned int* wcol  = (unsigned int*)(smem + SM_WCOL);

    const int c0   = blockIdx.x * CT;
    const int b    = blockIdx.y;
    const int tid  = threadIdx.x;
    const int lane = tid & 31;
    const int wid  = tid >> 5;
    const int n0w  = wid * 64;           // this wave's first region row
    const int m    = lane & 15;
    const int hi   = lane >> 4;          // 0: lanes 0-15, 1: lanes 16-31

    const float*     xb = x + (size_t)b * T_DIM * C_DIM + c0;
    const long long* rb = regions + (size_t)b * T_DIM;
    const unsigned short* sb_h = scale_h + b * N_DIM;

    v8f acc[4][4];
    #pragma unroll
    for (int i = 0; i < 4; ++i)
        #pragma unroll
        for (int j = 0; j < 4; ++j)
            acc[i][j] = (v8f)0.0f;

    // Prologue: kick off the first TDM stage.
    if (wid == 0) tdm_load_stage(SM_STAGE, xb);

    const int NSTAGE = T_DIM / KST;      // 32
    for (int s = 0; s < NSTAGE; ++s) {
        const int ks  = s * KST;
        const int buf = s & 1;
        if (wid == 0) __builtin_amdgcn_s_wait_tensorcnt(0);
        __syncthreads();                 // stage[buf] ready for everyone
        if (wid == 0 && (s + 1) < NSTAGE)
            tdm_load_stage(SM_STAGE + ((buf ^ 1) * (KST * CT * 4)),
                           xb + (size_t)(ks + KST) * C_DIM);

        // Pack (f16 scale | region-1) for this stage's 128 t-rows.
        if (tid < KST) {
            int r = (int)rb[ks + tid] - 1;
            unsigned int hb = (unsigned int)sb_h[r];
            wcol[tid] = (hb << 16) | (unsigned int)r;
        }
        // Convert f32 stage -> f16 [c][k] tile (32 elems/thread, coalesced reads).
        const float* sbuf = stage + buf * (KST * CT);
        #pragma unroll 4
        for (int e = tid; e < KST * CT; e += 256) {
            int k = e >> 6;              // row (t index within stage)
            int c = e & 63;              // column
            Xh[c * PITCH + k] = (_Float16)sbuf[e];
        }
        __syncthreads();                 // Xh + wcol ready

        // 4 K-chunks of 32 per stage.
        #pragma unroll
        for (int ch = 0; ch < 4; ++ch) {
            const int kc = ch * 32;
            // --- A fragments (one per 16-region subtile), built from wcol ---
            Frag16 A[4];
            const int kAbase = hi * 16;  // A layout: lanes<16 -> K 0..15, lanes>=16 -> 16..31
            #pragma unroll
            for (int j = 0; j < 16; ++j) {
                int kk = kAbase + j;
                unsigned int u = wcol[kc + kk];          // LDS broadcast
                int d = (int)(u & 0xFFFFu) - (n0w + m);
                unsigned short hb = (unsigned short)(u >> 16);
                A[0].u[j] = (d == 0)  ? hb : (unsigned short)0;
                A[1].u[j] = (d == 16) ? hb : (unsigned short)0;
                A[2].u[j] = (d == 32) ? hb : (unsigned short)0;
                A[3].u[j] = (d == 48) ? hb : (unsigned short)0;
            }
            // --- B fragments: 16 contiguous halves at fixed c ---
            const int kb = hi * 16;      // lanes<16 -> K 0..15, lanes>=16 -> 16..31
            Frag16 Bf[4];
            #pragma unroll
            for (int cs = 0; cs < 4; ++cs) {
                const _Float16* p = &Xh[(cs * 16 + m) * PITCH + kc + kb];
                Bf[cs].h[0] = *(const v8h*)p;
                Bf[cs].h[1] = *(const v8h*)(p + 8);
            }
            // --- 16 WMMAs ---
            #pragma unroll
            for (int i = 0; i < 4; ++i)
                #pragma unroll
                for (int j = 0; j < 4; ++j)
                    acc[i][j] = __builtin_amdgcn_wmma_f32_16x16x32_f16(
                        false, A[i].v, false, Bf[j].v,
                        (short)0, acc[i][j], false, false);
        }
        __syncthreads();                 // done reading Xh before next convert
    }

    // Store: D layout -> out[b][n][c], coalesced 64B per half-wave row.
    float* ob = out + (size_t)b * N_DIM * C_DIM;
    #pragma unroll
    for (int i = 0; i < 4; ++i) {
        int nbase = n0w + i * 16 + 8 * hi;
        #pragma unroll
        for (int j = 0; j < 4; ++j) {
            int c = c0 + j * 16 + m;
            #pragma unroll
            for (int r = 0; r < 8; ++r)
                ob[(size_t)(nbase + r) * C_DIM + c] = acc[i][j][r];
        }
    }
}

// ---------------------------------------------------------------------------
extern "C" void kernel_launch(void* const* d_in, const int* in_sizes, int n_in,
                              void* d_out, int out_size, void* d_ws, size_t ws_size,
                              hipStream_t stream)
{
    (void)in_sizes; (void)n_in; (void)out_size; (void)ws_size;
    const float*     x       = (const float*)d_in[0];
    const long long* regions = (const long long*)d_in[1];
    // d_in[2] = max_n scalar (fixed at 512 by the reference)
    unsigned short* scale_h = (unsigned short*)d_ws;   // 8*512*2 B scratch
    float* out = (float*)d_out;

    seg_count_kernel<<<dim3(B_DIM), dim3(256), 0, stream>>>(regions, scale_h);
    seg_gemm_kernel<<<dim3(C_DIM / CT, B_DIM), dim3(256), SM_TOTAL, stream>>>(
        x, regions, scale_h, out);
}
